// TransformerBlockWithALiBi_49091476193567
// MI455X (gfx1250) — matrix operand order repaired and checked by reading, not compiled, not run on hardware
//
#include <hip/hip_runtime.h>
#include <hip/hip_bf16.h>
#include <math.h>

typedef __bf16 bf16_t;
typedef __attribute__((ext_vector_type(16))) __bf16 v16bf;
typedef __attribute__((ext_vector_type(8)))  float  v8f;

#define DM     1024
#define NHEADS 16
#define HD     64
#define FFN    4096
#define SEQT   2048
#define LN_EPS 1e-5f

union U16bf { uint4 u[2]; v16bf v; };

// CDNA5 async global->LDS copy (ASYNCcnt-tracked), GV addressing mode.
// VDST = per-lane LDS byte offset, VADDR = per-lane 64-bit global address.
__device__ inline void async_load_b128(uint32_t lds_off, const void* g) {
  asm volatile("global_load_async_to_lds_b128 %0, %1, off"
               :: "v"(lds_off), "v"(g) : "memory");
}
__device__ inline void wait_async_le4() {
  asm volatile("s_wait_asynccnt 0x4" ::: "memory");
}
__device__ inline void wait_async_0() {
  asm volatile("s_wait_asynccnt 0x0" ::: "memory");
}
__device__ inline uint32_t lds_off_u32(const void* p) {
  return (uint32_t)(uintptr_t)p;   // flat LDS address: group-segment offset in [31:0]
}

// ---------------------------------------------------------------------------
// fp32 [K,N] -> bf16 [N,K]  (weight convert + transpose, tiled through LDS)
// grid: (N/32, K/32), block: 256
// ---------------------------------------------------------------------------
__global__ __launch_bounds__(256)
void convert_transpose_bf16(const float* __restrict__ w, bf16_t* __restrict__ wt,
                            int K, int N) {
  __shared__ float tile[32][33];
  const int kb = blockIdx.y * 32, nb = blockIdx.x * 32;
  const int tx = threadIdx.x & 31, ty = threadIdx.x >> 5;   // 32 x 8
  for (int i = 0; i < 32; i += 8)
    tile[ty + i][tx] = w[(size_t)(kb + ty + i) * N + nb + tx];
  __syncthreads();
  for (int i = 0; i < 32; i += 8)
    wt[(size_t)(nb + ty + i) * K + kb + tx] = (bf16_t)tile[tx][ty + i];
}

// ---------------------------------------------------------------------------
// LayerNorm: fp32 [T,DM] row -> bf16 [T,DM] row.  One block (256 thr) per row.
// ---------------------------------------------------------------------------
__global__ __launch_bounds__(256)
void layernorm_bf16(const float* __restrict__ x, const float* __restrict__ g,
                    const float* __restrict__ b, bf16_t* __restrict__ out) {
  __shared__ float red[256];
  const int row = blockIdx.x, tid = threadIdx.x;
  const float* xr = x + (size_t)row * DM;
  float local[4]; float s = 0.f;
  for (int i = 0; i < 4; ++i) { local[i] = xr[tid + i * 256]; s += local[i]; }
  red[tid] = s; __syncthreads();
  for (int st = 128; st > 0; st >>= 1) {
    if (tid < st) red[tid] += red[tid + st];
    __syncthreads();
  }
  const float mu = red[0] * (1.f / DM);
  __syncthreads();
  float s2 = 0.f;
  for (int i = 0; i < 4; ++i) { float d = local[i] - mu; s2 += d * d; }
  red[tid] = s2; __syncthreads();
  for (int st = 128; st > 0; st >>= 1) {
    if (tid < st) red[tid] += red[tid + st];
    __syncthreads();
  }
  const float rstd = rsqrtf(red[0] * (1.f / DM) + LN_EPS);
  for (int i = 0; i < 4; ++i) {
    const int c = tid + i * 256;
    out[(size_t)row * DM + c] = (bf16_t)((local[i] - mu) * rstd * g[c] + b[c]);
  }
}

// ---------------------------------------------------------------------------
// WMMA bf16 GEMM:  out = act(A[M,K] @ Bt[N,K]^T + bias[N]) (+ res[M,N])
// OUTMODE: 0 = fp32 [M,N], 1 = bf16 [M,N], 2 = bf16 transposed [N,M]
// ACT:     0 = none, 1 = exact GELU
// Tile 128x128x32, 256 threads = 8 waves, wave -> 32x64 C (2x4 WMMA tiles).
// Tiles staged with double-buffered GLOBAL_LOAD_ASYNC_TO_LDS_B128:
// next tile's HBM traffic overlaps the current tile's 8 WMMAs.
// ---------------------------------------------------------------------------
template <int OUTMODE, int ACT, bool RES>
__global__ __launch_bounds__(256)
void gemm_wmma(const bf16_t* __restrict__ A, const bf16_t* __restrict__ Bt,
               const float* __restrict__ bias, const float* __restrict__ res,
               void* __restrict__ out, int M, int N, int K) {
  constexpr int BM = 128, BN = 128, BK = 32;
  __shared__ bf16_t sA[2][BM * BK];
  __shared__ bf16_t sB[2][BN * BK];

  const int tid  = threadIdx.x;
  const int lane = tid & 31;
  const int wave = tid >> 5;
  const int wm   = wave >> 1;        // 0..3 (M dir)
  const int wn   = wave & 1;         // 0..1 (N dir)
  const int col  = lane & 15;
  const int hf   = lane >> 4;
  const int bm0  = blockIdx.y * BM;
  const int bn0  = blockIdx.x * BN;

  const v8f zero = {};
  v8f acc[2][4];
  for (int i = 0; i < 2; ++i)
    for (int j = 0; j < 4; ++j) acc[i][j] = zero;

  const int ldRow = tid >> 1;          // 0..127
  const int ldSeg = (tid & 1) * 16;    // 0 or 16 (bf16 elems)
  const bf16_t* gA = A  + (size_t)(bm0 + ldRow) * K + ldSeg;
  const bf16_t* gB = Bt + (size_t)(bn0 + ldRow) * K + ldSeg;
  const uint32_t ldsOff = (uint32_t)(ldRow * BK + ldSeg);

  // issue one stage of async tile copies (4 x b128 per lane -> ASYNCcnt += 4)
  auto stage = [&](int buf, int k0) {
    async_load_b128(lds_off_u32(&sA[buf][ldsOff]),     gA + k0);
    async_load_b128(lds_off_u32(&sA[buf][ldsOff + 8]), gA + k0 + 8);
    async_load_b128(lds_off_u32(&sB[buf][ldsOff]),     gB + k0);
    async_load_b128(lds_off_u32(&sB[buf][ldsOff + 8]), gB + k0 + 8);
  };

  stage(0, 0);
  int buf = 0;
  for (int k0 = 0; k0 < K; k0 += BK) {
    if (k0 + BK < K) {             // prefetch next tile, then retire current
      stage(buf ^ 1, k0 + BK);
      wait_async_le4();            // async loads complete in order
    } else {
      wait_async_0();
    }
    __syncthreads();

    v16bf afr[2], bfr[4];
    for (int mt = 0; mt < 2; ++mt) {           // A-frag: 16 contiguous K
      const bf16_t* p = &sA[buf][(wm * 32 + mt * 16 + col) * BK + (hf ? 16 : 0)];
      U16bf u; u.u[0] = *(const uint4*)p; u.u[1] = *(const uint4*)(p + 8);
      afr[mt] = u.v;
    }
    for (int nt = 0; nt < 4; ++nt) {           // B-frag: 16 contiguous K
      const bf16_t* p = &sB[buf][(wn * 64 + nt * 16 + col) * BK + (hf ? 16 : 0)];
      U16bf u; u.u[0] = *(const uint4*)p; u.u[1] = *(const uint4*)(p + 8);
      bfr[nt] = u.v;
    }
    for (int mt = 0; mt < 2; ++mt)
      for (int nt = 0; nt < 4; ++nt)
        acc[mt][nt] = __builtin_amdgcn_wmma_f32_16x16x32_bf16(
            false, afr[mt], false, bfr[nt], (short)0, acc[mt][nt], false, false);
    __syncthreads();               // all waves done reading buf before it is re-staged
    buf ^= 1;
  }

  for (int mt = 0; mt < 2; ++mt)
    for (int nt = 0; nt < 4; ++nt) {
      const int colg = bn0 + wn * 64 + nt * 16 + col;
      const float bv = bias[colg];
      for (int r = 0; r < 8; ++r) {
        const int rowg = bm0 + wm * 32 + mt * 16 + r + hf * 8;
        float v = acc[mt][nt][r] + bv;
        if (ACT == 1) v = 0.5f * v * (1.0f + erff(v * 0.70710678118654752f));
        if (RES) v += res[(size_t)rowg * N + colg];
        if (OUTMODE == 0)      ((float*)out)[(size_t)rowg * N + colg]  = v;
        else if (OUTMODE == 1) ((bf16_t*)out)[(size_t)rowg * N + colg] = (bf16_t)v;
        else                   ((bf16_t*)out)[(size_t)colg * M + rowg] = (bf16_t)v;
      }
    }
}

// ---------------------------------------------------------------------------
// Flash attention with analytic ALiBi + causal mask.
// One 32-lane wave per (head, 16-query block).  Inner step = 32 keys:
//   S = Q(16x64) Kblk^T  -> 4 WMMAs;  O += P(16x32) V(32x64) -> 4 WMMAs.
// Q,K: bf16 [T,DM];  Vt: bf16 [DM,T] (transposed);  Oa: bf16 [T,DM].
// ---------------------------------------------------------------------------
__global__ __launch_bounds__(32)
void attention_flash(const bf16_t* __restrict__ Q, const bf16_t* __restrict__ Kb,
                     const bf16_t* __restrict__ Vt, bf16_t* __restrict__ Oa) {
  const int lane = threadIdx.x;
  const int col  = lane & 15;
  const int hf   = lane >> 4;
  const int q0   = blockIdx.x * 16;
  const int h    = blockIdx.y;
  const float slope = exp2f(-0.5f * (float)(h + 1));
  const int   fbase = h * HD;

  __shared__ bf16_t pbuf[16 * 32];

  // Q A-fragments (K-dim = head features, split 0..31 / 32..63)
  v16bf qa[2];
  for (int f = 0; f < 2; ++f) {
    const bf16_t* p = Q + (size_t)(q0 + col) * DM + fbase + f * 32 + (hf ? 16 : 0);
    U16bf u; u.u[0] = *(const uint4*)p; u.u[1] = *(const uint4*)(p + 8);
    qa[f] = u.v;
  }

  const v8f zero = {};
  float m[8], l[8]; v8f o[4];
  for (int r = 0; r < 8; ++r) { m[r] = -INFINITY; l[r] = 0.f; }
  for (int t = 0; t < 4; ++t) o[t] = zero;

  const int jmax = (q0 + 15) >> 5;
  for (int j = 0; j <= jmax; ++j) {
    const int kb0 = j * 32;

    // ---- S = Q K^T for two 16-key sub-blocks ----
    v8f s[2];
    for (int sb = 0; sb < 2; ++sb) {
      const bf16_t* kp = Kb + (size_t)(kb0 + sb * 16 + col) * DM + fbase;
      v16bf kf0, kf1;
      { const bf16_t* p = kp + (hf ? 16 : 0);
        U16bf u; u.u[0] = *(const uint4*)p; u.u[1] = *(const uint4*)(p + 8); kf0 = u.v; }
      { const bf16_t* p = kp + 32 + (hf ? 16 : 0);
        U16bf u; u.u[0] = *(const uint4*)p; u.u[1] = *(const uint4*)(p + 8); kf1 = u.v; }
      v8f sa = __builtin_amdgcn_wmma_f32_16x16x32_bf16(false, qa[0], false, kf0,
                                                       (short)0, zero, false, false);
      sa = __builtin_amdgcn_wmma_f32_16x16x32_bf16(false, qa[1], false, kf1,
                                                   (short)0, sa, false, false);
      s[sb] = sa;
    }

    // ---- scale + ALiBi + causal mask ----
    float pv[2][8];
    for (int sb = 0; sb < 2; ++sb) {
      const int kpos = kb0 + sb * 16 + col;
      for (int r = 0; r < 8; ++r) {
        const int qpos = q0 + r + hf * 8;
        pv[sb][r] = (kpos > qpos) ? -INFINITY
                  : s[sb][r] * 0.125f + slope * (float)(kpos - qpos);
      }
    }

    // ---- online softmax update (row = 16 lanes of one half) ----
    float scaleo[8];
    for (int r = 0; r < 8; ++r) {
      float lm = fmaxf(pv[0][r], pv[1][r]);
      for (int msk = 8; msk >= 1; msk >>= 1) lm = fmaxf(lm, __shfl_xor(lm, msk, 32));
      const float mn = fmaxf(m[r], lm);
      scaleo[r] = expf(m[r] - mn);
      float rs = 0.f;
      for (int sb = 0; sb < 2; ++sb) {
        const float p = expf(pv[sb][r] - mn);
        pv[sb][r] = p; rs += p;
      }
      for (int msk = 8; msk >= 1; msk >>= 1) rs += __shfl_xor(rs, msk, 32);
      l[r] = l[r] * scaleo[r] + rs;
      m[r] = mn;
    }
    for (int t = 0; t < 4; ++t)
      for (int r = 0; r < 8; ++r) o[t][r] *= scaleo[r];

    // ---- transpose P (C-layout) -> A-fragment layout through LDS ----
    __syncthreads();
    for (int sb = 0; sb < 2; ++sb)
      for (int r = 0; r < 8; ++r)
        pbuf[(r + hf * 8) * 32 + sb * 16 + col] = (bf16_t)pv[sb][r];
    __syncthreads();
    v16bf pa;
    { const bf16_t* p = &pbuf[col * 32 + (hf ? 16 : 0)];
      U16bf u; u.u[0] = *(const uint4*)p; u.u[1] = *(const uint4*)(p + 8); pa = u.v; }

    // ---- O += P V  (V transposed: contiguous along keys) ----
    for (int nt = 0; nt < 4; ++nt) {
      const bf16_t* p = Vt + (size_t)(fbase + nt * 16 + col) * SEQT + kb0 + (hf ? 16 : 0);
      U16bf u; u.u[0] = *(const uint4*)p; u.u[1] = *(const uint4*)(p + 8);
      o[nt] = __builtin_amdgcn_wmma_f32_16x16x32_bf16(false, pa, false, u.v,
                                                      (short)0, o[nt], false, false);
    }
  }

  for (int nt = 0; nt < 4; ++nt)
    for (int r = 0; r < 8; ++r) {
      const float v = o[nt][r] / l[r];
      Oa[(size_t)(q0 + r + hf * 8) * DM + fbase + nt * 16 + col] = (bf16_t)v;
    }
}

// ---------------------------------------------------------------------------
extern "C" void kernel_launch(void* const* d_in, const int* in_sizes, int n_in,
                              void* d_out, int out_size, void* d_ws, size_t ws_size,
                              hipStream_t stream) {
  const float* x   = (const float*)d_in[0];
  // d_in[1] causal_mask, d_in[2] alibi_bias: computed analytically, not read.
  const float* wq  = (const float*)d_in[3];  const float* bq  = (const float*)d_in[4];
  const float* wk  = (const float*)d_in[5];  const float* bk  = (const float*)d_in[6];
  const float* wv  = (const float*)d_in[7];  const float* bv  = (const float*)d_in[8];
  const float* wo  = (const float*)d_in[9];  const float* bo  = (const float*)d_in[10];
  const float* w1  = (const float*)d_in[11]; const float* b1  = (const float*)d_in[12];
  const float* w2  = (const float*)d_in[13]; const float* b2  = (const float*)d_in[14];
  const float* g1  = (const float*)d_in[15]; const float* be1 = (const float*)d_in[16];
  const float* g2  = (const float*)d_in[17]; const float* be2 = (const float*)d_in[18];
  float* out = (float*)d_out;

  // workspace carve (~72 MB)
  char* ws = (char*)d_ws;
  auto carve = [&](size_t bytes) {
    char* p = ws; ws += (bytes + 255) & ~(size_t)255; return p;
  };
  bf16_t* wq_t = (bf16_t*)carve((size_t)DM * DM * 2);
  bf16_t* wk_t = (bf16_t*)carve((size_t)DM * DM * 2);
  bf16_t* wv_t = (bf16_t*)carve((size_t)DM * DM * 2);
  bf16_t* wo_t = (bf16_t*)carve((size_t)DM * DM * 2);
  bf16_t* w1_t = (bf16_t*)carve((size_t)DM * FFN * 2);   // [FFN][DM]
  bf16_t* w2_t = (bf16_t*)carve((size_t)FFN * DM * 2);   // [DM][FFN]
  bf16_t* xn1  = (bf16_t*)carve((size_t)SEQT * DM * 2);
  bf16_t* Qb   = (bf16_t*)carve((size_t)SEQT * DM * 2);
  bf16_t* Kbuf = (bf16_t*)carve((size_t)SEQT * DM * 2);
  bf16_t* Vt   = (bf16_t*)carve((size_t)DM * SEQT * 2);  // transposed [DM][T]
  bf16_t* attn = (bf16_t*)carve((size_t)SEQT * DM * 2);
  float*  x2   = (float*) carve((size_t)SEQT * DM * 4);
  bf16_t* xn2  = (bf16_t*)carve((size_t)SEQT * DM * 2);
  bf16_t* hbuf = (bf16_t*)carve((size_t)SEQT * FFN * 2);

  const dim3 blk256(256);
  // weight convert + transpose to bf16 [N][K]
  convert_transpose_bf16<<<dim3(DM / 32, DM / 32),  blk256, 0, stream>>>(wq, wq_t, DM, DM);
  convert_transpose_bf16<<<dim3(DM / 32, DM / 32),  blk256, 0, stream>>>(wk, wk_t, DM, DM);
  convert_transpose_bf16<<<dim3(DM / 32, DM / 32),  blk256, 0, stream>>>(wv, wv_t, DM, DM);
  convert_transpose_bf16<<<dim3(DM / 32, DM / 32),  blk256, 0, stream>>>(wo, wo_t, DM, DM);
  convert_transpose_bf16<<<dim3(FFN / 32, DM / 32), blk256, 0, stream>>>(w1, w1_t, DM, FFN);
  convert_transpose_bf16<<<dim3(DM / 32, FFN / 32), blk256, 0, stream>>>(w2, w2_t, FFN, DM);

  // LN1
  layernorm_bf16<<<SEQT, blk256, 0, stream>>>(x, g1, be1, xn1);

  // Q, K (bf16 [T,DM]) and V (bf16 transposed [DM,T])
  const dim3 gD(DM / 128, SEQT / 128);
  gemm_wmma<1, 0, false><<<gD, blk256, 0, stream>>>(xn1, wq_t, bq, nullptr, Qb,   SEQT, DM, DM);
  gemm_wmma<1, 0, false><<<gD, blk256, 0, stream>>>(xn1, wk_t, bk, nullptr, Kbuf, SEQT, DM, DM);
  gemm_wmma<2, 0, false><<<gD, blk256, 0, stream>>>(xn1, wv_t, bv, nullptr, Vt,   SEQT, DM, DM);

  // flash attention with analytic ALiBi + causal
  attention_flash<<<dim3(SEQT / 16, NHEADS), dim3(32), 0, stream>>>(Qb, Kbuf, Vt, attn);

  // O projection + residual -> x2 (fp32)
  gemm_wmma<0, 0, true><<<gD, blk256, 0, stream>>>(attn, wo_t, bo, x, x2, SEQT, DM, DM);

  // LN2
  layernorm_bf16<<<SEQT, blk256, 0, stream>>>(x2, g2, be2, xn2);

  // FFN1 + exact GELU -> h (bf16)
  gemm_wmma<1, 1, false><<<dim3(FFN / 128, SEQT / 128), blk256, 0, stream>>>(
      xn2, w1_t, b1, nullptr, hbuf, SEQT, FFN, DM);

  // FFN2 + residual -> out (fp32)
  gemm_wmma<0, 0, true><<<gD, blk256, 0, stream>>>(hbuf, w2_t, b2, x2, out, SEQT, DM, FFN);
}
